// HomographySmoothLoss_8529805050140
// MI455X (gfx1250) — matrix operand order repaired and checked by reading, not compiled, not run on hardware
//
#include <hip/hip_runtime.h>
#include <math.h>

// ---------------------------------------------------------------------------
// HomographySmoothLoss (per-segment affine flow fit) for MI455X / gfx1250.
//
// Memory-bound problem (96 MB streamed, ~4 us at 23.3 TB/s). The segment
// scatter-reduction is done as a one-hot matmul on the WMMA path
// (v_wmma_f32_16x16x32_f16) instead of conflicted LDS atomics:
//   C[16 segs][16 stats] += onehot[16x32] @ stats[32x16]
// Coordinates are normalized to [0,1) so f16 stat inputs are well-scaled;
// the loss is invariant under this affine reparametrization. Accumulation
// is f32 inside the WMMA, so pixel counts are exact.
//
// Round-2 change: all four one-hot A tiles are materialized before the four
// WMMAs issue, so the WMMAs go back-to-back (distinct A/D, shared B -> no
// hazard) instead of each being followed by 4 v_nops for the WAR-on-A
// hazard against the next one-hot's v_cndmask writes.
// ---------------------------------------------------------------------------

typedef __attribute__((ext_vector_type(16))) _Float16 v16h;
typedef __attribute__((ext_vector_type(8)))  float    v8f;

#define NUM_SEGS 64
#define BATCH    8
#define HGT      512
#define WID      1024
#define PIX_PER_BATCH (HGT * WID)                       // 524288
#define BLOCKS_PER_BATCH 64
#define CHUNK (PIX_PER_BATCH / BLOCKS_PER_BATCH)        // 8192 pixels / block
#define ITERS (CHUNK / 256)                             // 32 (8 waves x 32 px)
#define G_TOTAL (BATCH * NUM_SEGS)                      // 512 global segments
#define NSTATS 16                                       // 12 used, padded to 16

// workspace layout (in floats)
#define WS_STATS  0
#define WS_PARAMS (G_TOTAL * NSTATS)                    // 8192
#define WS_RESSUM (WS_PARAMS + G_TOTAL * 8)             // 12288
#define WS_FLOATS (WS_RESSUM + G_TOTAL)                 // 12800 floats = 50 KB

// ---------------------------------------------------------------------------
__global__ void hsl_zero_kernel(float* p, int n) {
    for (int i = blockIdx.x * blockDim.x + threadIdx.x; i < n;
         i += gridDim.x * blockDim.x)
        p[i] = 0.0f;
}

// ---------------------------------------------------------------------------
// Pass 1: per-segment sufficient statistics via one-hot WMMA.
// grid = BATCH * BLOCKS_PER_BATCH blocks, 256 threads (8 wave32).
// Stats per pixel (normalized coords xn=x/W, yn=y/H):
//   0:xn^2 1:xn*yn 2:xn 3:yn^2 4:yn 5:1 6:xn*u 7:yn*u 8:u 9:xn*v 10:yn*v 11:v
// ---------------------------------------------------------------------------
__global__ __launch_bounds__(256)
void hsl_stats_kernel(const float* __restrict__ flow,
                      const int*   __restrict__ masks,
                      float*       __restrict__ gstats) {
    const int b    = blockIdx.x / BLOCKS_PER_BATCH;
    const int blk  = blockIdx.x % BLOCKS_PER_BATCH;
    const int tid  = threadIdx.x;
    const int wave = tid >> 5;
    const int lane = tid & 31;

    // stat-major staging tile: B-operand read is one contiguous 32B per lane
    __shared__ __align__(32) _Float16 tile[8][NSTATS][32];   // 8 KB
    __shared__ int   segs[8][32];                            // 1 KB
    __shared__ float lbins[NUM_SEGS * NSTATS];               // 4 KB

    // zero padded stat rows (12..15) once, and the block-local bins
    for (int i = tid; i < 8 * NSTATS * 32; i += 256)
        ((_Float16*)tile)[i] = (_Float16)0.0f;
    for (int i = tid; i < NUM_SEGS * NSTATS; i += 256)
        lbins[i] = 0.0f;
    __syncthreads();

    const float invW = 1.0f / (float)WID;
    const float invH = 1.0f / (float)HGT;
    const float* uptr = flow + (size_t)b * 2 * PIX_PER_BATCH;
    const float* vptr = uptr + PIX_PER_BATCH;
    const int*   mptr = masks + (size_t)b * PIX_PER_BATCH;
    const int    base = blk * CHUNK;

    v8f acc[4] = {};   // 4 x (16 segs x 16 stats) f32 accumulators = 64 segs

    const int mrow = lane & 15;            // C/A row within 16-seg group
    const int kbA  = (lane < 16) ? 0 : 16; // A-operand K base for this lane
    const int kbB  = (lane < 16) ? 0 : 16; // B-operand K base for this lane

    for (int it = 0; it < ITERS; ++it) {
        const int pix = base + ((it * 8 + wave) << 5) + lane; // coalesced
        const float u = uptr[pix];
        const float v = vptr[pix];
        const int   m = mptr[pix];
        const float xn = (float)(pix & (WID - 1)) * invW;
        const float yn = (float)(pix >> 10) * invH;

        __syncthreads();  // WAR: previous iteration's reads done
        tile[wave][0][lane]  = (_Float16)(xn * xn);
        tile[wave][1][lane]  = (_Float16)(xn * yn);
        tile[wave][2][lane]  = (_Float16)xn;
        tile[wave][3][lane]  = (_Float16)(yn * yn);
        tile[wave][4][lane]  = (_Float16)yn;
        tile[wave][5][lane]  = (_Float16)1.0f;
        tile[wave][6][lane]  = (_Float16)(xn * u);
        tile[wave][7][lane]  = (_Float16)(yn * u);
        tile[wave][8][lane]  = (_Float16)u;
        tile[wave][9][lane]  = (_Float16)(xn * v);
        tile[wave][10][lane] = (_Float16)(yn * v);
        tile[wave][11][lane] = (_Float16)v;
        segs[wave][lane] = m;
        __syncthreads();  // RAW: staged data visible

        // B operand (32x16 stats, K x N): lane holds column n = lane&15,
        // K = kbB..kbB+15 -> contiguous 16 halves in stat-major tile.
        const v16h bmat =
            *reinterpret_cast<const v16h*>(&tile[wave][mrow][kbB]);

        // Segment ids for this lane's A-operand K slots.
        int sv[16];
#pragma unroll
        for (int slot = 0; slot < 16; ++slot) {
            const int k = kbA + slot;
            sv[slot] = segs[wave][k];
        }

        // Materialize all 4 one-hot A tiles first ...
        v16h amat[4];
#pragma unroll
        for (int g = 0; g < 4; ++g) {
            const int want = (g << 4) + mrow;
#pragma unroll
            for (int slot = 0; slot < 16; ++slot)
                amat[g][slot] = (sv[slot] == want) ? (_Float16)1.0f
                                                   : (_Float16)0.0f;
        }
        // ... then issue the 4 WMMAs back-to-back (no A/B/D overlap between
        // consecutive WMMAs -> no hazard NOPs in between).
#pragma unroll
        for (int g = 0; g < 4; ++g)
            acc[g] = __builtin_amdgcn_wmma_f32_16x16x32_f16(
                false, amat[g], false, bmat, (short)0, acc[g], false, false);
    }

    // Reduce the 8 waves' C tiles into block-local bins, then to global.
    __syncthreads();
#pragma unroll
    for (int g = 0; g < 4; ++g) {
#pragma unroll
        for (int r = 0; r < 8; ++r) {
            const int M = r + ((lane < 16) ? 0 : 8);       // C row
            const int idx = ((g << 4) + M) * NSTATS + mrow; // seg*16 + stat
            atomicAdd(&lbins[idx], acc[g][r]);
        }
    }
    __syncthreads();
    float* gdst = gstats + (size_t)b * NUM_SEGS * NSTATS;
    for (int i = tid; i < NUM_SEGS * NSTATS; i += 256)
        atomicAdd(&gdst[i], lbins[i]);
}

// ---------------------------------------------------------------------------
// Pass 2: 512 symmetric 3x3 solves (Cramer, f64). params[g*8 + 0..5],
// [6]=valid flag, [7]=count.
// ---------------------------------------------------------------------------
__global__ void hsl_solve_kernel(const float* __restrict__ gstats,
                                 float* __restrict__ params) {
    const int g = blockIdx.x * blockDim.x + threadIdx.x;
    if (g >= G_TOTAL) return;
    const float* s = gstats + g * NSTATS;
    const float cnt = s[5];
    const bool valid = (cnt >= 100.0f) && ((g & (NUM_SEGS - 1)) != 0);

    float out[8] = {0.f, 0.f, 0.f, 0.f, 0.f, 0.f, 0.f, cnt};
    if (valid) {
        const double a = s[0], bq = s[1], c = s[2];
        const double d = s[3], e = s[4], f = s[5];
        const double A00 = d * f - e * e;
        const double A01 = c * e - bq * f;
        const double A02 = bq * e - d * c;
        const double A11 = a * f - c * c;
        const double A12 = bq * c - a * e;
        const double A22 = a * d - bq * bq;
        const double det = a * A00 + bq * A01 + c * A02;
        const double inv = (det != 0.0) ? 1.0 / det : 0.0;
        {
            const double r0 = s[6], r1 = s[7], r2 = s[8];
            out[0] = (float)((A00 * r0 + A01 * r1 + A02 * r2) * inv);
            out[1] = (float)((A01 * r0 + A11 * r1 + A12 * r2) * inv);
            out[2] = (float)((A02 * r0 + A12 * r1 + A22 * r2) * inv);
        }
        {
            const double r0 = s[9], r1 = s[10], r2 = s[11];
            out[3] = (float)((A00 * r0 + A01 * r1 + A02 * r2) * inv);
            out[4] = (float)((A01 * r0 + A11 * r1 + A12 * r2) * inv);
            out[5] = (float)((A02 * r0 + A12 * r1 + A22 * r2) * inv);
        }
        out[6] = 1.0f;
    }
#pragma unroll
    for (int i = 0; i < 8; ++i) params[g * 8 + i] = out[i];
}

// ---------------------------------------------------------------------------
// Pass 3: residual norms, per-segment sums. Param table (16 KB) is
// cache-resident; one LDS atomic per pixel into 64 bins.
// ---------------------------------------------------------------------------
__global__ __launch_bounds__(256)
void hsl_residual_kernel(const float* __restrict__ flow,
                         const int*   __restrict__ masks,
                         const float* __restrict__ params,
                         float*       __restrict__ res_sum) {
    const int b   = blockIdx.x / BLOCKS_PER_BATCH;
    const int blk = blockIdx.x % BLOCKS_PER_BATCH;
    const int tid = threadIdx.x;

    __shared__ float bins[NUM_SEGS];
    if (tid < NUM_SEGS) bins[tid] = 0.0f;
    __syncthreads();

    const float invW = 1.0f / (float)WID;
    const float invH = 1.0f / (float)HGT;
    const float* uptr = flow + (size_t)b * 2 * PIX_PER_BATCH;
    const float* vptr = uptr + PIX_PER_BATCH;
    const int*   mptr = masks + (size_t)b * PIX_PER_BATCH;
    const int    base = blk * CHUNK;
    const float* ptab = params + (size_t)b * NUM_SEGS * 8;

    for (int i = tid; i < CHUNK; i += 256) {
        const int pix = base + i;
        const float u = uptr[pix];
        const float v = vptr[pix];
        const int   m = mptr[pix];
        const float xn = (float)(pix & (WID - 1)) * invW;
        const float yn = (float)(pix >> 10) * invH;
        const float* p = ptab + (m << 3);
        const float du = u - (p[0] * xn + p[1] * yn + p[2]);
        const float dv = v - (p[3] * xn + p[4] * yn + p[5]);
        atomicAdd(&bins[m], sqrtf(du * du + dv * dv));
    }
    __syncthreads();
    if (tid < NUM_SEGS) atomicAdd(&res_sum[b * NUM_SEGS + tid], bins[tid]);
}

// ---------------------------------------------------------------------------
// Pass 4: loss = mean over valid segments of (res_sum / max(cnt,1)).
// ---------------------------------------------------------------------------
__global__ void hsl_finalize_kernel(const float* __restrict__ params,
                                    const float* __restrict__ res_sum,
                                    float* __restrict__ out) {
    const int tid = threadIdx.x;  // 512 threads
    __shared__ float ssum[G_TOTAL];
    __shared__ int   scnt[G_TOTAL];
    float contrib = 0.0f;
    int   vcount  = 0;
    const float* p = params + tid * 8;
    if (p[6] != 0.0f) {
        vcount = 1;
        contrib = res_sum[tid] / fmaxf(p[7], 1.0f);
    }
    ssum[tid] = contrib;
    scnt[tid] = vcount;
    __syncthreads();
    for (int s = G_TOTAL / 2; s > 0; s >>= 1) {
        if (tid < s) { ssum[tid] += ssum[tid + s]; scnt[tid] += scnt[tid + s]; }
        __syncthreads();
    }
    if (tid == 0)
        out[0] = (scnt[0] > 0) ? ssum[0] / (float)scnt[0] : 0.0f;
}

// ---------------------------------------------------------------------------
extern "C" void kernel_launch(void* const* d_in, const int* in_sizes, int n_in,
                              void* d_out, int out_size, void* d_ws,
                              size_t ws_size, hipStream_t stream) {
    (void)in_sizes; (void)n_in; (void)out_size; (void)ws_size;
    const float* flow  = (const float*)d_in[0];  // [8,2,512,1024] f32
    const int*   masks = (const int*)d_in[1];    // [8,512,1024]   i32

    float* ws      = (float*)d_ws;
    float* gstats  = ws + WS_STATS;    // [512][16]
    float* params  = ws + WS_PARAMS;   // [512][8]
    float* res_sum = ws + WS_RESSUM;   // [512]
    float* out     = (float*)d_out;

    hsl_zero_kernel<<<50, 256, 0, stream>>>(ws, WS_FLOATS);
    hsl_stats_kernel<<<BATCH * BLOCKS_PER_BATCH, 256, 0, stream>>>(
        flow, masks, gstats);
    hsl_solve_kernel<<<2, 256, 0, stream>>>(gstats, params);
    hsl_residual_kernel<<<BATCH * BLOCKS_PER_BATCH, 256, 0, stream>>>(
        flow, masks, params, res_sum);
    hsl_finalize_kernel<<<1, G_TOTAL, 0, stream>>>(params, res_sum, out);
}
